// SurrealEmbedding_56650618634407
// MI455X (gfx1250) — matrix operand order repaired and checked by reading, not compiled, not run on hardware
//
#include <hip/hip_runtime.h>
#include <hip/hip_bf16.h>
#include <math.h>

// ---------------------------------------------------------------------------
// SurrealEmbedding: hv = Wplus @ base_plus + Wminus @ base_minus, row-normalized.
// Weights generated on the fly from signs/lengths. GEMM on
// v_wmma_f32_16x16x32_f16 (wave32, gfx1250). LDS tiles are stored in the exact
// byte layout the WMMA fragment registers need, so fragments are assembled with
// raw ds_load_b128 (no VALU repacking). Staging is fully branchless and uses
// 32-bit element indices so global accesses lower to GVS mode
// (SGPR base + 32-bit voffset) instead of per-lane 64-bit address math.
// ---------------------------------------------------------------------------

typedef __attribute__((ext_vector_type(16))) _Float16 v16h;
typedef __attribute__((ext_vector_type(8)))  _Float16 v8h;
typedef __attribute__((ext_vector_type(8)))  float    v8f;

#define BM 128            // rows (batch) per workgroup
#define BN 128            // cols (D) per workgroup
#define BK 32             // K-step (matches 16x16x32 f16 WMMA)
#define ASTR 40           // row stride (halfs) of weight tiles: 80B rows (16B aligned), 20-dword stride -> conflict-free
#define BSTR 40           // row stride (halfs) of transposed base tiles: same property
#define NTHREADS 256      // 8 wave32s

static constexpr float ALPHA_F = 0.6180339887498949f;  // 1/phi
static constexpr float BETA_F  = 0.3819660112501051f;  // 1/phi^2
static constexpr float SUMW_F  = ALPHA_F + BETA_F;     // == 1.0f

__device__ __forceinline__ unsigned hbits(_Float16 h) {
    union { _Float16 h; unsigned short u; } x; x.h = h; return (unsigned)x.u;
}
__device__ __forceinline__ unsigned packf2(float a, float b) {
    union { _Float16 h[2]; unsigned u; } x;
    x.h[0] = (_Float16)a; x.h[1] = (_Float16)b; return x.u;
}

// Concatenate two 16-byte LDS loads into a fragment register block.
__device__ __forceinline__ v16h frag_cat(const v8h* p0, const v8h* p1) {
    v8h a = *p0, b = *p1;
    return __builtin_shufflevector(a, b, 0, 1, 2, 3, 4, 5, 6, 7,
                                         8, 9, 10, 11, 12, 13, 14, 15);
}

__global__ void __launch_bounds__(NTHREADS)
surreal_gemm_wmma(const int* __restrict__ signs,     // [B, L]
                  const int* __restrict__ lengths,   // [B]
                  const float* __restrict__ base_p,  // [L, D]
                  const float* __restrict__ base_m,  // [L, D]
                  float* __restrict__ out,           // [B, D] (un-normalized here)
                  int B, int L, int D)
{
    __shared__ int      nArr[BM];
    __shared__ _Float16 wpT[BM * ASTR];   // A (plus)  tile: [row][k], f16, fragment-native layout
    __shared__ _Float16 wmT[BM * ASTR];   // A (minus) tile
    __shared__ _Float16 pT[BN * BSTR];    // base_plus  tile, transposed [n][k]
    __shared__ _Float16 mT[BN * BSTR];    // base_minus tile, transposed [n][k]

    const int tid  = threadIdx.x;
    const int lane = tid & 31;
    const int lm   = lane & 15;         // M (A/C) or N (B/C) index within fragment
    const int hs   = lane >> 4;         // upper half-wave selector

    const int wave = tid >> 5;          // 0..7
    const int wm   = wave >> 2;         // 0..1  -> 64 rows each
    const int wn   = wave & 3;          // 0..3  -> 32 cols each

    const int row0 = blockIdx.y * BM;
    const int n0   = blockIdx.x * BN;

    if (tid < BM) {
        const int gb = row0 + tid;
        nArr[tid] = (gb < B) ? lengths[min(gb, B - 1)] : 0;
    }

    v8f acc[4][2];
#pragma unroll
    for (int f = 0; f < 4; ++f)
#pragma unroll
        for (int g = 0; g < 2; ++g)
            acc[f][g] = v8f{0.f, 0.f, 0.f, 0.f, 0.f, 0.f, 0.f, 0.f};

    const unsigned wAlphaBits = hbits((_Float16)ALPHA_F);
    const unsigned wSumBits   = hbits((_Float16)SUMW_F);

    for (int k0 = 0; k0 < L; k0 += BK) {
        __syncthreads();   // protect LDS from previous iteration's readers (and nArr, iter 0)

        // ---- Stage A weights (branchless): two k's per thread, b32 stores ---
        // All element indices fit int32 (B*L ~5.1M) -> GVS addressing.
#pragma unroll
        for (int i = tid; i < BM * (BK / 2); i += NTHREADS) {
            const int row = i >> 4;               // BK/2 == 16
            const int kk2 = (i & 15) * 2;
            const int gb  = row0 + row;
            const int n   = nArr[row];
            const int sb  = min(gb, B - 1) * L;   // int32 row offset

            const int l0  = k0 + kk2;
            const int l1  = l0 + 1;
            const int s0  = signs[sb + min(l0, L - 1)];
            const int s1  = signs[sb + min(l1, L - 1)];

            const bool in0 = (gb < B) & (l0 < L) & (l0 < n);
            const bool in1 = (gb < B) & (l1 < L) & (l1 < n);
            const unsigned w0 = in0 ? ((l0 == 0) ? wAlphaBits : wSumBits) : 0u;
            const unsigned w1 = in1 ? ((l1 == 0) ? wAlphaBits : wSumBits) : 0u;

            const unsigned wp  = (s0 == 0 ? w0 : 0u) | ((s1 == 0 ? w1 : 0u) << 16);
            const unsigned wm2 = (s0 != 0 ? w0 : 0u) | ((s1 != 0 ? w1 : 0u) << 16);
            *(unsigned*)(wpT + row * ASTR + kk2) = wp;
            *(unsigned*)(wmT + row * ASTR + kk2) = wm2;
        }

        // ---- Stage f16 base tiles (branchless, transposed [n][k]) -----------
        // Indices fit int32 (L*D ~1.2M) -> GVS addressing.
#pragma unroll
        for (int i = tid; i < (BK / 2) * BN; i += NTHREADS) {
            const int nn  = i & (BN - 1);         // consecutive tid -> consecutive n (coalesced)
            const int kk  = (i >> 7) * 2;         // BN == 128
            const int gn  = n0 + nn;
            const int gk0 = k0 + kk;
            const int gnc = min(gn, D - 1);
            const int i0  = min(gk0, L - 1) * D + gnc;
            const int i1  = min(gk0 + 1, L - 1) * D + gnc;

            float p0 = base_p[i0];
            float p1 = base_p[i1];
            float m0 = base_m[i0];
            float m1 = base_m[i1];

            const bool okn = (gn < D);
            const bool ok0 = okn & (gk0 < L);
            const bool ok1 = okn & (gk0 + 1 < L);
            p0 = ok0 ? p0 : 0.f;  m0 = ok0 ? m0 : 0.f;
            p1 = ok1 ? p1 : 0.f;  m1 = ok1 ? m1 : 0.f;

            *(unsigned*)(pT + nn * BSTR + kk) = packf2(p0, p1);
            *(unsigned*)(mT + nn * BSTR + kk) = packf2(m0, m1);
        }

        __syncthreads();

        // ---- B fragments: 16 contiguous halfs per lane (2x b128, no repack) --
        v16h bp[2], bm[2];
        const int kOffB = hs * 16;                // upper lanes hold K+16
#pragma unroll
        for (int g = 0; g < 2; ++g) {
            const int nBase = wn * 32 + g * 16 + lm;
            const v8h* pr = (const v8h*)(pT + nBase * BSTR + kOffB);
            const v8h* mr = (const v8h*)(mT + nBase * BSTR + kOffB);
            bp[g] = frag_cat(pr, pr + 1);
            bm[g] = frag_cat(mr, mr + 1);
        }

        // ---- A fragments: 16 contiguous halfs [kSel..kSel+15] ----------------
        const int kSelA = hs * 16;                // upper lanes hold K+16
#pragma unroll
        for (int f = 0; f < 4; ++f) {
            const int mBase = wm * 64 + f * 16 + lm;
            const v8h* ar = (const v8h*)(wpT + mBase * ASTR + kSelA);
            const v8h* br = (const v8h*)(wmT + mBase * ASTR + kSelA);
            const v16h ap = frag_cat(ar, ar + 1); // +1 v8h == +8 halfs
            const v16h am = frag_cat(br, br + 1);
#pragma unroll
            for (int g = 0; g < 2; ++g) {
                acc[f][g] = __builtin_amdgcn_wmma_f32_16x16x32_f16(
                    false, ap, false, bp[g], (short)0, acc[f][g], false, false);
                acc[f][g] = __builtin_amdgcn_wmma_f32_16x16x32_f16(
                    false, am, false, bm[g], (short)0, acc[f][g], false, false);
            }
        }
    }

    // ---- Store C (16x16 f32 layout: VGPR r -> M = r + hs*8, N = lane&15) ----
    // out indices fit int32 (B*D ~15.6M) -> GVS addressing.
    const bool fullM = (row0 + BM <= B);          // uniform fast path
#pragma unroll
    for (int f = 0; f < 4; ++f) {
#pragma unroll
        for (int g = 0; g < 2; ++g) {
            const int nGlob = n0 + wn * 32 + g * 16 + lm;
            const int mBase = row0 + wm * 64 + f * 16 + hs * 8;
            if (nGlob < D) {
                const int o0 = mBase * D + nGlob;
                if (fullM) {
#pragma unroll
                    for (int r = 0; r < 8; ++r)
                        out[o0 + r * D] = acc[f][g][r];
                } else {
#pragma unroll
                    for (int r = 0; r < 8; ++r)
                        if (mBase + r < B)
                            out[o0 + r * D] = acc[f][g][r];
                }
            }
        }
    }
}

// ---------------------------------------------------------------------------
// In-place row L2 normalization.
// ---------------------------------------------------------------------------
__global__ void __launch_bounds__(NTHREADS)
surreal_rownorm(float* __restrict__ out, int D)
{
    __shared__ float red[NTHREADS];
    const int row = blockIdx.x;
    const int tid = threadIdx.x;
    float* rowp = out + row * D;   // int32 offset: B*D fits

    float s = 0.f;
    for (int d = tid; d < D; d += NTHREADS) {
        const float v = rowp[d];
        s += v * v;
    }
    red[tid] = s;
    __syncthreads();
#pragma unroll
    for (int off = NTHREADS / 2; off > 0; off >>= 1) {
        if (tid < off) red[tid] += red[tid + off];
        __syncthreads();
    }
    if (tid == 0) {
        const float nrm = sqrtf(red[0]);
        red[0] = (nrm > 0.f) ? (1.f / nrm) : 1.f;
    }
    __syncthreads();
    const float inv = red[0];
    for (int d = tid; d < D; d += NTHREADS)
        rowp[d] *= inv;
}

extern "C" void kernel_launch(void* const* d_in, const int* in_sizes, int n_in,
                              void* d_out, int out_size, void* d_ws, size_t ws_size,
                              hipStream_t stream)
{
    const int*   signs   = (const int*)d_in[0];    // [B, L]
    const int*   lengths = (const int*)d_in[1];    // [B]
    const float* base_p  = (const float*)d_in[2];  // [L, D]
    const float* base_m  = (const float*)d_in[3];  // [L, D]
    float*       out     = (float*)d_out;          // [B, D]

    const int B = in_sizes[1];
    const int L = in_sizes[0] / B;
    const int D = in_sizes[2] / L;

    dim3 grid((D + BN - 1) / BN, (B + BM - 1) / BM);
    surreal_gemm_wmma<<<grid, NTHREADS, 0, stream>>>(signs, lengths, base_p, base_m,
                                                     out, B, L, D);
    surreal_rownorm<<<B, NTHREADS, 0, stream>>>(out, D);
}
